// MyModel_15676630630731
// MI455X (gfx1250) — matrix operand order repaired and checked by reading, not compiled, not run on hardware
//
#include <hip/hip_runtime.h>
#include <hip/hip_bf16.h>

#define BATCH  64
#define MAXLEN 256
#define EMBD   300
#define EPAD   320
#define HID    512
#define G4     2048
#define NCLS   10

typedef __attribute__((ext_vector_type(8)))  float          v8f;
typedef __attribute__((ext_vector_type(16))) __bf16         v16bf;
typedef __attribute__((ext_vector_type(8)))  unsigned short u16x8;
typedef __attribute__((ext_vector_type(16))) unsigned short u16x16;

union BFrag { u16x16 u16; u16x8 u8[2]; v16bf bf; };

__device__ inline unsigned short f2bf(float f) {
  unsigned u = __float_as_uint(f);
  u += 0x7FFFu + ((u >> 16) & 1u);           // round-to-nearest-even
  return (unsigned short)(u >> 16);
}
__device__ inline float sigmoidf_(float x) { return 1.0f / (1.0f + __expf(-x)); }

// A fragment (16x32 bf16, MxK): lanes 0-15 -> row M=lane,    K = k0+0..7  & k0+16..23
//                               lanes 16-31-> row M=lane-16, K = k0+8..15 & k0+24..31
// caller passes pointer already offset by row*stride + k0 + (lane<16?0:8)
__device__ inline v16bf load_a(const unsigned short* p) {
  BFrag f;
  f.u8[0] = *(const u16x8*)(p);
  f.u8[1] = *(const u16x8*)(p + 16);
  return f.bf;
}
// B fragment (32x16 bf16, KxN) from N-major (transposed) weights:
// lane reads K = k0+0..7 & k0+16..23 (lanes 0-15) / k0+8..15 & k0+24..31 (lanes 16-31)
// of its column; caller offsets by n_row*stride + k0 + (lane<16?0:8)
__device__ inline v16bf load_b(const unsigned short* p) {
  BFrag f;
  f.u8[0] = *(const u16x8*)(p);
  f.u8[1] = *(const u16x8*)(p + 16);
  return f.bf;
}

// ---------------------------------------------------------------- init
__global__ void init_kernel(unsigned* cnt, unsigned* hbuf_u32, int n) {
  int i = blockIdx.x * blockDim.x + threadIdx.x;
  if (i == 0) *cnt = 0u;
  for (; i < n; i += gridDim.x * blockDim.x) hbuf_u32[i] = 0u;
}

// ------------------------------------------------- weight transpose+cast
// WxT[d][n][k] (k padded to 320), WhT[d][n][k]
__global__ void prep_weights_kernel(const float* __restrict__ Wx_f,
                                    const float* __restrict__ Wh_f,
                                    const float* __restrict__ Wx_b,
                                    const float* __restrict__ Wh_b,
                                    unsigned short* __restrict__ WxT,
                                    unsigned short* __restrict__ WhT) {
  const int NX = 2 * G4 * EPAD;        // 1310720
  const int NH = G4 * HID;             // 1048576
  const int total = NX + 2 * NH;
  for (int i = blockIdx.x * blockDim.x + threadIdx.x; i < total;
       i += gridDim.x * blockDim.x) {
    if (i < NX) {
      int d   = i / (G4 * EPAD);
      int rem = i % (G4 * EPAD);
      int n   = rem / EPAD;
      int k   = rem % EPAD;
      const float* W = d ? Wx_b : Wx_f;          // [300][2048] row-major
      float v = (k < EMBD) ? W[k * G4 + n] : 0.0f;
      WxT[i] = f2bf(v);
    } else {
      int j   = i - NX;
      int d   = j / NH;
      int rem = j % NH;
      int n   = rem / HID;
      int k   = rem % HID;
      const float* W = d ? Wh_b : Wh_f;          // [512][2048] row-major
      WhT[j] = f2bf(W[k * G4 + n]);
    }
  }
}

// ------------------------------------------------- embedding gather+cast
// xe[t][b][e], e padded to 320
__global__ void gather_kernel(const int* __restrict__ x,
                              const float* __restrict__ emb,
                              unsigned short* __restrict__ xe) {
  const int total = MAXLEN * BATCH * EPAD;
  for (int i = blockIdx.x * blockDim.x + threadIdx.x; i < total;
       i += gridDim.x * blockDim.x) {
    int e = i % EPAD;
    int b = (i / EPAD) % BATCH;
    int t = i / (EPAD * BATCH);
    int tok = x[b * MAXLEN + t];
    float v = (e < EMBD) ? emb[(size_t)tok * EMBD + e] : 0.0f;
    xe[i] = f2bf(v);
  }
}

// ------------------------------------------------- persistent BiLSTM
// 32 blocks x 256 threads = 256 waves.
// wave -> (dir, m_tile in [0,4), j_tile in [0,32)); owns 16x16 (batch,hid) patch,
// computes all 4 gate tiles each step; cell state c stays in VGPRs for all 256 steps.
// All 4 waves sharing a B fragment (same j_tile, different m_tile) live in the same
// block -> weight re-reads are WGP$-local; unique L2 weight traffic/step = full
// bf16 weight set once (~6.6 MB), which is L2-resident (192 MB).
__global__ void __launch_bounds__(256)
lstm_kernel(const unsigned short* __restrict__ xe,
            const unsigned short* __restrict__ WxT,
            const unsigned short* __restrict__ WhT,
            const float* __restrict__ b_f, const float* __restrict__ b_b,
            unsigned short* __restrict__ hbuf,   // [dir][2][64][512] bf16
            float* __restrict__ hfinal,          // [64][1024] f32
            unsigned* __restrict__ cnt) {
  const int lane  = threadIdx.x & 31;
  const int gwave = blockIdx.x * (blockDim.x >> 5) + (threadIdx.x >> 5); // 0..255
  const int dir   = gwave >> 7;
  const int rid   = gwave & 127;
  const int m0    = (rid & 3) << 4;     // batch tile origin
  const int j0    = (rid >> 2) << 4;    // hidden tile origin
  const int l15   = lane & 15;
  const int aoff  = (lane < 16) ? 0 : 8;
  const int boff  = (lane < 16) ? 0 : 8;
  const int nrow  = j0 + l15;           // this lane's N (hidden) column
  const int arow  = m0 + l15;           // this lane's A row (batch)

  const float* bias = dir ? b_b : b_f;
  const float bv0 = bias[0 * HID + nrow];
  const float bv1 = bias[1 * HID + nrow];
  const float bv2 = bias[2 * HID + nrow];
  const float bv3 = bias[3 * HID + nrow];

  const unsigned short* wxt0 = WxT + (size_t)dir * G4 * EPAD;
  const unsigned short* wht0 = WhT + (size_t)dir * G4 * HID;

  v8f cst = {};                         // cell state tile, register-resident
  for (int step = 0; step < MAXLEN; ++step) {
    const int t = dir ? (MAXLEN - 1 - step) : step;

    // Defeat LICM without losing the global address space: launder a zero
    // 32-bit offset through an opaque asm and add it to the (addrspace(1)-
    // inferred) weight pointers.  The loads stay in-loop (no 830-VGPR spill)
    // AND still lower to global_load_b128, not flat_load_b128 (flat would
    // burn DScnt/LDS-issue bandwidth and aperture checks).
    unsigned zx = 0u, zh = 0u;
    asm volatile("" : "+v"(zx), "+v"(zh));
    const unsigned short* wxt = wxt0 + zx;
    const unsigned short* wht = wht0 + zh;

    v8f acc0 = {}, acc1 = {}, acc2 = {}, acc3 = {};

    // ---- fused input projection: K = 320 (bf16 WMMA, f32 accum)
    const unsigned short* xet = xe + ((size_t)t * BATCH + arow) * EPAD;
    for (int k0 = 0; k0 < EPAD; k0 += 32) {
      v16bf a = load_a(xet + k0 + aoff);
      acc0 = __builtin_amdgcn_wmma_f32_16x16x32_bf16(false, a, false,
              load_b(wxt + (size_t)(0 * HID + nrow) * EPAD + k0 + boff),
              (short)0, acc0, false, false);
      acc1 = __builtin_amdgcn_wmma_f32_16x16x32_bf16(false, a, false,
              load_b(wxt + (size_t)(1 * HID + nrow) * EPAD + k0 + boff),
              (short)0, acc1, false, false);
      acc2 = __builtin_amdgcn_wmma_f32_16x16x32_bf16(false, a, false,
              load_b(wxt + (size_t)(2 * HID + nrow) * EPAD + k0 + boff),
              (short)0, acc2, false, false);
      acc3 = __builtin_amdgcn_wmma_f32_16x16x32_bf16(false, a, false,
              load_b(wxt + (size_t)(3 * HID + nrow) * EPAD + k0 + boff),
              (short)0, acc3, false, false);
    }

    // ---- recurrent projection: h_prev @ Wh, K = 512
    const unsigned short* hp =
        hbuf + (((size_t)(dir * 2 + (step & 1)) * BATCH) + arow) * HID;
    for (int k0 = 0; k0 < HID; k0 += 32) {
      v16bf a = load_a(hp + k0 + aoff);
      acc0 = __builtin_amdgcn_wmma_f32_16x16x32_bf16(false, a, false,
              load_b(wht + (size_t)(0 * HID + nrow) * HID + k0 + boff),
              (short)0, acc0, false, false);
      acc1 = __builtin_amdgcn_wmma_f32_16x16x32_bf16(false, a, false,
              load_b(wht + (size_t)(1 * HID + nrow) * HID + k0 + boff),
              (short)0, acc1, false, false);
      acc2 = __builtin_amdgcn_wmma_f32_16x16x32_bf16(false, a, false,
              load_b(wht + (size_t)(2 * HID + nrow) * HID + k0 + boff),
              (short)0, acc2, false, false);
      acc3 = __builtin_amdgcn_wmma_f32_16x16x32_bf16(false, a, false,
              load_b(wht + (size_t)(3 * HID + nrow) * HID + k0 + boff),
              (short)0, acc3, false, false);
    }

    // ---- gate nonlinearities + state update (Keras order i,f,g,o; relu act)
    unsigned short* hn =
        hbuf + ((size_t)(dir * 2 + ((step + 1) & 1)) * BATCH) * HID;
#pragma unroll
    for (int q = 0; q < 8; ++q) {
      float ii = sigmoidf_(acc0[q] + bv0);
      float ff = sigmoidf_(acc1[q] + bv1);
      float gg = fmaxf(acc2[q] + bv2, 0.0f);
      float oo = sigmoidf_(acc3[q] + bv3);
      float c  = ff * cst[q] + ii * gg;
      cst[q] = c;
      float h = oo * fmaxf(c, 0.0f);
      // C/D layout: VGPR q -> M = m0+q (lanes 0-15) / m0+8+q (lanes 16-31)
      const int m = m0 + q + ((lane < 16) ? 0 : 8);
      hn[(size_t)m * HID + nrow] = f2bf(h);
      if (step == MAXLEN - 1)
        hfinal[(size_t)m * (2 * HID) + dir * HID + nrow] = h;
    }

    // ---- grid-wide step barrier (agent-scope release/acquire)
    __threadfence();
    __syncthreads();
    if (threadIdx.x == 0) {
      __hip_atomic_fetch_add(cnt, 1u, __ATOMIC_RELEASE, __HIP_MEMORY_SCOPE_AGENT);
      const unsigned target = (unsigned)gridDim.x * (unsigned)(step + 1);
      while (__hip_atomic_load(cnt, __ATOMIC_ACQUIRE, __HIP_MEMORY_SCOPE_AGENT) <
             target) {
        __builtin_amdgcn_s_sleep(8);
      }
    }
    __syncthreads();
    __threadfence();
  }
}

// ------------------------------------------------- dense + softmax
__global__ void dense_softmax_kernel(const float* __restrict__ hfinal,
                                     const float* __restrict__ Wd,
                                     const float* __restrict__ bd,
                                     float* __restrict__ out) {
  const int row  = blockIdx.x;
  const int lane = threadIdx.x;   // blockDim = 32 (one wave)
  const float* hr = hfinal + (size_t)row * (2 * HID);
  float logits[NCLS];
#pragma unroll
  for (int c = 0; c < NCLS; ++c) {
    float s = 0.0f;
    for (int k = lane; k < 2 * HID; k += 32) s += hr[k] * Wd[(size_t)k * NCLS + c];
    for (int o = 16; o > 0; o >>= 1) s += __shfl_down(s, o, 32);
    logits[c] = s;
  }
  if (lane == 0) {
    float mx = -1e30f;
#pragma unroll
    for (int c = 0; c < NCLS; ++c) { logits[c] += bd[c]; mx = fmaxf(mx, logits[c]); }
    float den = 0.0f;
#pragma unroll
    for (int c = 0; c < NCLS; ++c) { logits[c] = __expf(logits[c] - mx); den += logits[c]; }
    const float inv = 1.0f / den;
#pragma unroll
    for (int c = 0; c < NCLS; ++c) out[row * NCLS + c] = logits[c] * inv;
  }
}

// ------------------------------------------------- launcher
extern "C" void kernel_launch(void* const* d_in, const int* in_sizes, int n_in,
                              void* d_out, int out_size, void* d_ws, size_t ws_size,
                              hipStream_t stream) {
  const int*   x    = (const int*)  d_in[0];
  const float* emb  = (const float*)d_in[1];
  const float* Wx_f = (const float*)d_in[2];
  const float* Wh_f = (const float*)d_in[3];
  const float* b_f  = (const float*)d_in[4];
  const float* Wx_b = (const float*)d_in[5];
  const float* Wh_b = (const float*)d_in[6];
  const float* b_b  = (const float*)d_in[7];
  const float* Wd   = (const float*)d_in[8];
  const float* bd   = (const float*)d_in[9];
  float* out = (float*)d_out;

  char* ws = (char*)d_ws;
  size_t off = 0;
  unsigned*       cnt    = (unsigned*)(ws);              off += 256;
  unsigned short* hbuf   = (unsigned short*)(ws + off);  off += (size_t)2 * 2 * BATCH * HID * 2;
  float*          hfinal = (float*)(ws + off);           off += (size_t)BATCH * 2 * HID * 4;
  unsigned short* WxT    = (unsigned short*)(ws + off);  off += (size_t)2 * G4 * EPAD * 2;
  unsigned short* WhT    = (unsigned short*)(ws + off);  off += (size_t)2 * G4 * HID * 2;
  unsigned short* xe     = (unsigned short*)(ws + off);  off += (size_t)MAXLEN * BATCH * EPAD * 2;

  init_kernel<<<256, 256, 0, stream>>>(cnt, (unsigned*)hbuf,
                                       (2 * 2 * BATCH * HID * 2) / 4);
  prep_weights_kernel<<<2048, 256, 0, stream>>>(Wx_f, Wh_f, Wx_b, Wh_b, WxT, WhT);
  gather_kernel<<<4096, 256, 0, stream>>>(x, emb, xe);
  lstm_kernel<<<32, 256, 0, stream>>>(xe, WxT, WhT, b_f, b_b, hbuf, hfinal, cnt);
  dense_softmax_kernel<<<BATCH, 32, 0, stream>>>(hfinal, Wd, bd, out);
}
